// GraphEncoder_37864431681715
// MI455X (gfx1250) — matrix operand order repaired and checked by reading, not compiled, not run on hardware
//
#include <hip/hip_runtime.h>
#include <hip/hip_bf16.h>

// ---------------------------------------------------------------------------
// Problem constants (GraphEncoder reference)
// ---------------------------------------------------------------------------
#define BB     2
#define SS     512
#define HH     768
#define NHEADS 12
#define HD     64
#define MROWS  (BB * SS)          // 1024
#define KTOP   (SS / 2)           // 256
#define ETOT   (BB * SS * KTOP)   // 262144 edges
#define LN_EPS 1e-5f

// d_out layout: node_features | edge_index | edge_attr
#define OUT_NODE_OFF 0
#define OUT_EI_OFF   (MROWS * HH)                 // 786432
#define OUT_EA_OFF   (OUT_EI_OFF + 2 * ETOT)      // 1310720

// Workspace layout (byte offsets, all 256B aligned)
#define WS_HS16   0u
#define WS_WP16   (WS_HS16  + (size_t)MROWS * HH * 2)   // 1572864
#define WS_WQ16   (WS_WP16  + (size_t)HH * HH * 2)
#define WS_WK16   (WS_WQ16  + (size_t)HH * HH * 2)
#define WS_NODE32 (WS_WK16  + (size_t)HH * HH * 2)
#define WS_NODE16 (WS_NODE32 + (size_t)MROWS * HH * 4)
#define WS_Q16    (WS_NODE16 + (size_t)MROWS * HH * 2)
#define WS_K16    (WS_Q16    + (size_t)MROWS * HH * 2)
#define WS_AVG    (WS_K16    + (size_t)MROWS * HH * 2)
#define WS_SCORES (WS_AVG    + (size_t)MROWS * SS * 4)
#define WS_WE0    (WS_SCORES + (size_t)ETOT * 4)

typedef __attribute__((ext_vector_type(16))) _Float16 v16h;
typedef __attribute__((ext_vector_type(8)))  _Float16 v8h;
typedef __attribute__((ext_vector_type(8)))  float    v8f;
typedef __attribute__((ext_vector_type(4)))  float    f32x4;

// ---------------------------------------------------------------------------
// Fragment loaders (wave32 WMMA, contiguous-k order chosen by this code, 16-bit operands, K=32 step)
//  A (16x32 MxK): lane L<16 -> row m0+L, 16 contiguous halves at k0
//                 lane L>=16 -> row m0+L-16, 16 contiguous halves at k0+16
//  B (32x16 KxN as W[N,K] row-major): lane L<16 -> row n0+L, 16 contiguous
//                 halves at k0 ; lanes 16..31 at k0+16  (one 32B load)
// ---------------------------------------------------------------------------
__device__ __forceinline__ v16h load_a_frag(const _Float16* __restrict__ p) {
  return *(const v16h*)p;
}
__device__ __forceinline__ v16h load_b_frag(const _Float16* __restrict__ p) {
  return *(const v16h*)p;
}

// ---------------------------------------------------------------------------
// f32 -> f16 conversion (grid-stride)
// ---------------------------------------------------------------------------
__global__ __launch_bounds__(256) void cvt_f16_kernel(
    const float* __restrict__ src, _Float16* __restrict__ dst, int n) {
  int stride = gridDim.x * blockDim.x;
  for (int i = blockIdx.x * blockDim.x + threadIdx.x; i < n; i += stride)
    dst[i] = (_Float16)src[i];
}

// ---------------------------------------------------------------------------
// WMMA GEMM:  C[m,n] = sum_k A[m,k] * W[n,k] + bias[n]
//   A: [M,K] f16 row-major, W: [N,K] f16 row-major.
//   Optional f32 output Cf and/or f16 output Ch.
//   One wave computes one 16x16 tile; block = 8 waves.
// ---------------------------------------------------------------------------
__global__ __launch_bounds__(256) void wmma_gemm_kernel(
    const _Float16* __restrict__ A, const _Float16* __restrict__ W,
    const float* __restrict__ bias, float* __restrict__ Cf,
    _Float16* __restrict__ Ch, int M, int N, int K) {
  const int lane = threadIdx.x;        // 0..31
  const int wave = threadIdx.y;        // 0..7
  const int ntn  = N >> 4;
  const int tile = blockIdx.x * 8 + wave;
  if (tile >= (M >> 4) * ntn) return;  // wave-uniform
  const int m0 = (tile / ntn) << 4;
  const int n0 = (tile % ntn) << 4;
  const int lh = lane & 15, lg = lane >> 4;

  const _Float16* pa = A + (size_t)(m0 + lh) * K + lg * 16;
  const _Float16* pb = W + (size_t)(n0 + lh) * K + lg * 16;

  v8f c = {};
  for (int k0 = 0; k0 < K; k0 += 32) {
    v16h af = load_a_frag(pa + k0);
    v16h bf = load_b_frag(pb + k0);
    __builtin_prefetch(pa + k0 + 32, 0, 1);
    __builtin_prefetch(pb + k0 + 32, 0, 1);
    c = __builtin_amdgcn_wmma_f32_16x16x32_f16(false, af, false, bf,
                                               (short)0, c, false, false);
  }

  const int col  = n0 + lh;
  const int row0 = m0 + lg * 8;
  const float bc = bias[col];
#pragma unroll
  for (int r = 0; r < 8; ++r) {
    float v = c[r] + bc;
    if (Cf) Cf[(size_t)(row0 + r) * N + col] = v;
    if (Ch) Ch[(size_t)(row0 + r) * N + col] = (_Float16)v;
  }
}

// ---------------------------------------------------------------------------
// Fused attention: for one (batch, 16-row q stripe):
//   for each head: logits(16x512) via WMMA -> LDS, masked softmax per row,
//   accumulate head-mean probabilities in registers; write avg [B*S, S].
// Block = 256 threads = 8 waves; wave w covers columns [w*64, w*64+64).
// Softmax mapping: thread t -> row t/16, 32-column segment (t%16)*32.
// ---------------------------------------------------------------------------
__global__ __launch_bounds__(256) void attn_avg_kernel(
    const _Float16* __restrict__ Q, const _Float16* __restrict__ Km,
    const int* __restrict__ amask, float* __restrict__ avg) {
  __shared__ float lgt[16][SS + 4];   // +4 pad: rows land in distinct banks
  __shared__ float sred[16][16];

  const int tid  = threadIdx.x;
  const int lane = tid & 31;
  const int wave = tid >> 5;
  const int b    = blockIdx.x >> 5;        // S/16 = 32 stripes per batch
  const int m0   = (blockIdx.x & 31) << 4; // q row within batch
  const int rowg = b * SS + m0;            // global q row
  const int lh = lane & 15, lg = lane >> 4;

  // softmax-phase mapping
  const int r  = tid >> 4;
  const int c0 = (tid & 15) * 32;
  unsigned mbits = 0;
#pragma unroll
  for (int i = 0; i < 32; ++i)
    mbits |= (unsigned)(amask[b * SS + c0 + i] != 0) << i;

  float acc[32];
#pragma unroll
  for (int i = 0; i < 32; ++i) acc[i] = 0.f;

  for (int h = 0; h < NHEADS; ++h) {
    const int kb = h * HD;
    // ---- logits stripe via WMMA (4 N-tiles per wave) ----
#pragma unroll
    for (int t = 0; t < 4; ++t) {
      const int n0 = (wave * 4 + t) << 4;
      const _Float16* pq = Q  + (size_t)(rowg + lh) * HH + kb + lg * 16;
      const _Float16* pk = Km + (size_t)(b * SS + n0 + lh) * HH + kb + lg * 16;
      v8f c = {};
      v16h a0 = load_a_frag(pq);
      v16h b0 = load_b_frag(pk);
      c = __builtin_amdgcn_wmma_f32_16x16x32_f16(false, a0, false, b0,
                                                 (short)0, c, false, false);
      v16h a1 = load_a_frag(pq + 32);
      v16h b1 = load_b_frag(pk + 32);
      c = __builtin_amdgcn_wmma_f32_16x16x32_f16(false, a1, false, b1,
                                                 (short)0, c, false, false);
      const int col = n0 + lh, rw0 = lg * 8;
#pragma unroll
      for (int rr = 0; rr < 8; ++rr)
        lgt[rw0 + rr][col] = c[rr] * 0.125f;   // / sqrt(64)
    }
    __syncthreads();

    // ---- masked softmax, 16 threads per row ----
    float vals[32], vmax = -3.0e38f;
#pragma unroll
    for (int i = 0; i < 32; ++i) {
      float v = lgt[r][c0 + i];
      v = ((mbits >> i) & 1u) ? v : -1e9f;
      vals[i] = v;
      vmax = fmaxf(vmax, v);
    }
    sred[r][tid & 15] = vmax;
    __syncthreads();
    float m = -3.0e38f;
#pragma unroll
    for (int j = 0; j < 16; ++j) m = fmaxf(m, sred[r][j]);
    __syncthreads();
    float lsum = 0.f;
#pragma unroll
    for (int i = 0; i < 32; ++i) { vals[i] = expf(vals[i] - m); lsum += vals[i]; }
    sred[r][tid & 15] = lsum;
    __syncthreads();
    float s = 0.f;
#pragma unroll
    for (int j = 0; j < 16; ++j) s += sred[r][j];
    const float inv = 1.f / s;
#pragma unroll
    for (int i = 0; i < 32; ++i) acc[i] += vals[i] * inv;
    __syncthreads();  // protect lgt before next head overwrites it
  }

  float* dst = avg + (size_t)(b * SS + m0 + r) * SS + c0;
#pragma unroll
  for (int i = 0; i < 32; ++i) dst[i] = acc[i] * (1.f / NHEADS);
}

// ---------------------------------------------------------------------------
// Top-k (k=256 of 512) per row, stable by index; emits edge_index (as f32
// into d_out) and gathered scores (ws). Rank-count select + prefix-sum
// compaction gives ascending-index order, matching the reference's sort.
// ---------------------------------------------------------------------------
__global__ __launch_bounds__(512) void topk_kernel(
    const float* __restrict__ avg, float* __restrict__ scores,
    float* __restrict__ out) {
  __shared__ float v[SS];
  __shared__ int   scan[SS];
  const int row  = blockIdx.x;       // 0..1023
  const int b    = row >> 9;
  const int srow = row & 511;
  const int tid  = threadIdx.x;

  const float x = avg[(size_t)row * SS + tid];
  v[tid] = x;
  __syncthreads();

  int rank = 0;
  for (int i = 0; i < SS; ++i) {
    float vi = v[i];
    rank += (vi > x) || (vi == x && i < tid);
  }
  const int sel = (rank < KTOP) ? 1 : 0;

  scan[tid] = sel;
  __syncthreads();
  for (int off = 1; off < SS; off <<= 1) {
    int add = (tid >= off) ? scan[tid - off] : 0;
    __syncthreads();
    scan[tid] += add;
    __syncthreads();
  }
  if (sel) {
    const int pos = scan[tid] - 1;            // exclusive position
    const int e   = row * KTOP + pos;
    out[OUT_EI_OFF + e]        = (float)(srow + b * SS);  // src + offset
    out[OUT_EI_OFF + ETOT + e] = (float)(tid + b * SS);   // dst + offset
    scores[e] = x;
  }
}

// ---------------------------------------------------------------------------
// Gather We[:,0] into a contiguous vector for the edge_attr stream.
// ---------------------------------------------------------------------------
__global__ __launch_bounds__(256) void we0_kernel(
    const float* __restrict__ We, float* __restrict__ we0) {
  int i = blockIdx.x * blockDim.x + threadIdx.x;
  if (i < HH) we0[i] = We[(size_t)i * HH];
}

// ---------------------------------------------------------------------------
// edge_attr[e][:] = score[e] * We[:,0] + be  — 805 MB pure store stream.
// One block per edge, 192 threads x float4, non-temporal stores.
// ---------------------------------------------------------------------------
__global__ __launch_bounds__(192) void edge_attr_kernel(
    const float* __restrict__ scores, const float* __restrict__ we0,
    const float* __restrict__ be, float* __restrict__ out_ea) {
  const int e = blockIdx.x;
  const int c = threadIdx.x;                 // 0..191 (float4 index)
  const float s = scores[e];
  const f32x4 w  = ((const f32x4*)we0)[c];
  const f32x4 bb = ((const f32x4*)be)[c];
  f32x4 r;
  r.x = s * w.x + bb.x;
  r.y = s * w.y + bb.y;
  r.z = s * w.z + bb.z;
  r.w = s * w.w + bb.w;
  __builtin_nontemporal_store(r, (f32x4*)out_ea + (size_t)e * (HH / 4) + c);
}

// ---------------------------------------------------------------------------
// LayerNorm over the feature dim (768 = 3 x 256) per row.
// ---------------------------------------------------------------------------
__global__ __launch_bounds__(256) void ln_kernel(
    const float* __restrict__ node, const float* __restrict__ gamma,
    const float* __restrict__ beta, float* __restrict__ out) {
  __shared__ float s1[256], s2[256];
  const int row = blockIdx.x;
  const int tid = threadIdx.x;
  const float* p = node + (size_t)row * HH;
  float a0 = p[tid], a1 = p[tid + 256], a2 = p[tid + 512];
  s1[tid] = a0 + a1 + a2;
  s2[tid] = a0 * a0 + a1 * a1 + a2 * a2;
  __syncthreads();
  for (int off = 128; off > 0; off >>= 1) {
    if (tid < off) { s1[tid] += s1[tid + off]; s2[tid] += s2[tid + off]; }
    __syncthreads();
  }
  const float mu  = s1[0] * (1.f / HH);
  const float var = s2[0] * (1.f / HH) - mu * mu;
  const float rs  = rsqrtf(var + LN_EPS);
  float* q = out + (size_t)row * HH;
  q[tid]       = (a0 - mu) * rs * gamma[tid]       + beta[tid];
  q[tid + 256] = (a1 - mu) * rs * gamma[tid + 256] + beta[tid + 256];
  q[tid + 512] = (a2 - mu) * rs * gamma[tid + 512] + beta[tid + 512];
}

// ---------------------------------------------------------------------------
extern "C" void kernel_launch(void* const* d_in, const int* in_sizes, int n_in,
                              void* d_out, int out_size, void* d_ws,
                              size_t ws_size, hipStream_t stream) {
  (void)in_sizes; (void)n_in; (void)out_size; (void)ws_size;
  const float* hs    = (const float*)d_in[0];
  const int*   amask = (const int*)d_in[1];
  const float* Wp    = (const float*)d_in[2];
  const float* bp    = (const float*)d_in[3];
  const float* Wq    = (const float*)d_in[4];
  const float* bq    = (const float*)d_in[5];
  const float* Wk    = (const float*)d_in[6];
  const float* bk    = (const float*)d_in[7];
  const float* We    = (const float*)d_in[8];
  const float* be    = (const float*)d_in[9];
  const float* gamma = (const float*)d_in[10];
  const float* beta  = (const float*)d_in[11];
  float* out = (float*)d_out;
  char*  ws  = (char*)d_ws;

  _Float16* hs16   = (_Float16*)(ws + WS_HS16);
  _Float16* wp16   = (_Float16*)(ws + WS_WP16);
  _Float16* wq16   = (_Float16*)(ws + WS_WQ16);
  _Float16* wk16   = (_Float16*)(ws + WS_WK16);
  float*    node32 = (float*)   (ws + WS_NODE32);
  _Float16* node16 = (_Float16*)(ws + WS_NODE16);
  _Float16* q16    = (_Float16*)(ws + WS_Q16);
  _Float16* k16    = (_Float16*)(ws + WS_K16);
  float*    avg    = (float*)   (ws + WS_AVG);
  float*    scores = (float*)   (ws + WS_SCORES);
  float*    we0    = (float*)   (ws + WS_WE0);

  // 1) f32 -> f16 conversions
  cvt_f16_kernel<<<1024, 256, 0, stream>>>(hs, hs16, MROWS * HH);
  cvt_f16_kernel<<<1024, 256, 0, stream>>>(Wp, wp16, HH * HH);
  cvt_f16_kernel<<<1024, 256, 0, stream>>>(Wq, wq16, HH * HH);
  cvt_f16_kernel<<<1024, 256, 0, stream>>>(Wk, wk16, HH * HH);

  // 2) node = hs @ Wp^T + bp  (f32 + f16 outputs)
  const dim3 gblk(32, 8);
  const int proj_tiles = (MROWS / 16) * (HH / 16);   // 3072 tiles / 8 waves
  wmma_gemm_kernel<<<proj_tiles / 8, gblk, 0, stream>>>(
      hs16, wp16, bp, node32, node16, MROWS, HH, HH);

  // 3) q/k projections (f16 outputs)
  wmma_gemm_kernel<<<proj_tiles / 8, gblk, 0, stream>>>(
      node16, wq16, bq, nullptr, q16, MROWS, HH, HH);
  wmma_gemm_kernel<<<proj_tiles / 8, gblk, 0, stream>>>(
      node16, wk16, bk, nullptr, k16, MROWS, HH, HH);

  // 4) fused attention logits + softmax + head-mean
  attn_avg_kernel<<<BB * (SS / 16), 256, 0, stream>>>(q16, k16, amask, avg);

  // 5) top-256 per row -> edge_index (f32) + scores
  topk_kernel<<<MROWS, SS, 0, stream>>>(avg, scores, out);

  // 6) We[:,0] gather, then 805 MB edge_attr store stream
  we0_kernel<<<(HH + 255) / 256, 256, 0, stream>>>(We, we0);
  edge_attr_kernel<<<ETOT, HH / 4 / 1, 0, stream>>>(scores, we0, be,
                                                    out + OUT_EA_OFF);

  // 7) LayerNorm -> node_features
  ln_kernel<<<MROWS, 256, 0, stream>>>(node32, gamma, beta, out);
}
